// NodeModel_7464653160946
// MI455X (gfx1250) — compile-verified, matrix operand order checked by reading, not run on hardware
//
#include <hip/hip_runtime.h>
#include <hip/hip_bf16.h>
#include <stdint.h>

typedef __attribute__((ext_vector_type(16))) __bf16 bf16x16;
typedef __attribute__((ext_vector_type(8)))  float  f32x8;

__device__ __forceinline__ f32x8 zero8() {
    f32x8 z;
#pragma unroll
    for (int i = 0; i < 8; ++i) z[i] = 0.f;
    return z;
}

__device__ __forceinline__ f32x8 wmma_bf16(bf16x16 a, bf16x16 b, f32x8 c) {
    return __builtin_amdgcn_wmma_f32_16x16x32_bf16(false, a, false, b, (short)0, c, false, false);
}

// A fragment (16x32 bf16) from a global fp32 row; p points at the 32-float K-tile.
// lanes 0-15 hold K {0..7,16..23}, lanes 16-31 hold K {8..15,24..31}, 2 per VGPR.
__device__ __forceinline__ bf16x16 make_afrag_g(const float* p, int half) {
    const float* q0 = p + 8 * half;
    const float* q1 = p + 16 + 8 * half;
    float4 r0 = *(const float4*)(q0);
    float4 r1 = *(const float4*)(q0 + 4);
    float4 r2 = *(const float4*)(q1);
    float4 r3 = *(const float4*)(q1 + 4);
    bf16x16 a;
    a[0]=(__bf16)r0.x; a[1]=(__bf16)r0.y; a[2]=(__bf16)r0.z; a[3]=(__bf16)r0.w;
    a[4]=(__bf16)r1.x; a[5]=(__bf16)r1.y; a[6]=(__bf16)r1.z; a[7]=(__bf16)r1.w;
    a[8]=(__bf16)r2.x; a[9]=(__bf16)r2.y; a[10]=(__bf16)r2.z; a[11]=(__bf16)r2.w;
    a[12]=(__bf16)r3.x; a[13]=(__bf16)r3.y; a[14]=(__bf16)r3.z; a[15]=(__bf16)r3.w;
    return a;
}

// A fragment from an LDS bf16 row (row-major, contiguous K).
__device__ __forceinline__ bf16x16 lds_afrag(const __bf16* row, int kt, int half) {
    union { int4 v[2]; bf16x16 b; } u;
    u.v[0] = *(const int4*)(row + kt * 32 + 8 * half);
    u.v[1] = *(const int4*)(row + kt * 32 + 16 + 8 * half);
    return u.b;
}

// Pre-swizzled B fragment: layout [frag][lane][8 dwords of packed bf16 pairs].
__device__ __forceinline__ bf16x16 load_bfrag(const uint32_t* Wp, int frag, int lane) {
    const uint32_t* q = Wp + ((size_t)frag * 32 + lane) * 8;
    union { int4 v[2]; bf16x16 b; } u;
    u.v[0] = *(const int4*)q;
    u.v[1] = *(const int4*)(q + 4);
    return u.b;
}

// ---------------- setup kernels ----------------

__global__ void zero_k(float* p, size_t n) {
    size_t i = (size_t)blockIdx.x * blockDim.x + threadIdx.x;
    size_t st = (size_t)gridDim.x * blockDim.x;
    for (; i < n; i += st) p[i] = 0.f;
}

// Convert fp32 weight [K][Nc] row-major into B-fragment packed layout.
__global__ void swizzle_w_k(const float* __restrict__ W, uint32_t* __restrict__ out,
                            int K, int Nc) {
    int total = (K / 32) * (Nc / 16) * 256;
    int idx = blockIdx.x * blockDim.x + threadIdx.x;
    if (idx >= total) return;
    int v = idx & 7;
    int lane = (idx >> 3) & 31;
    int f = idx >> 8;
    int NT = Nc / 16;
    int nt = f % NT;
    int kt = f / NT;
    int half = lane >> 4;
    int n = nt * 16 + (lane & 15);
    int kk = (v < 4) ? (8 * half + 2 * v) : (16 + 8 * half + 2 * (v - 4));
    int k = kt * 32 + kk;
    unsigned short lo = __builtin_bit_cast(unsigned short, (__bf16)W[(size_t)k * Nc + n]);
    unsigned short hi = __builtin_bit_cast(unsigned short, (__bf16)W[(size_t)(k + 1) * Nc + n]);
    out[idx] = ((uint32_t)hi << 16) | lo;
}

__global__ void finalize_k(const float* __restrict__ sum, const float* __restrict__ sumsq,
                           const float* __restrict__ g, const float* __restrict__ be,
                           float* __restrict__ scale, float* __restrict__ shift,
                           float invn, int C) {
    int c = blockIdx.x * blockDim.x + threadIdx.x;
    if (c >= C) return;
    float mu = sum[c] * invn;
    float var = sumsq[c] * invn - mu * mu;
    float sc = g[c] * rsqrtf(var + 1e-5f);
    scale[c] = sc;
    shift[c] = be[c] - mu * sc;
}

// ---------------- pass 1: edge BN statistics (bias-less GEMM1) ----------------

__global__ void __launch_bounds__(256) edge_stats_k(
    const float* __restrict__ x, const int* __restrict__ ei,
    const float* __restrict__ ea, const uint32_t* __restrict__ W1p,
    float* __restrict__ sum, float* __restrict__ sumsq, int E, int ntiles) {
    __shared__ __align__(16) float red[8][8][32][2];
    int lane = threadIdx.x & 31, wave = threadIdx.x >> 5;
    int half = lane >> 4, m = lane & 15;
    float s[8], ss[8];
#pragma unroll
    for (int nt = 0; nt < 8; ++nt) { s[nt] = 0.f; ss[nt] = 0.f; }
    int gw = blockIdx.x * 8 + wave, nw = gridDim.x * 8;
    for (int tile = gw; tile < ntiles; tile += nw) {
        int e = tile * 16 + m;
        int r = ei[e];
        const float* xr = x + (size_t)r * 64;
        const float* er = ea + (size_t)e * 64;
        f32x8 acc[8];
#pragma unroll
        for (int nt = 0; nt < 8; ++nt) acc[nt] = zero8();
#pragma unroll
        for (int kt = 0; kt < 4; ++kt) {
            const float* src = (kt < 2) ? (xr + kt * 32) : (er + (kt - 2) * 32);
            bf16x16 a = make_afrag_g(src, half);
#pragma unroll
            for (int nt = 0; nt < 8; ++nt) {
                bf16x16 b = load_bfrag(W1p, kt * 8 + nt, lane);
                acc[nt] = wmma_bf16(a, b, acc[nt]);
            }
        }
#pragma unroll
        for (int nt = 0; nt < 8; ++nt)
#pragma unroll
            for (int v = 0; v < 8; ++v) { float c = acc[nt][v]; s[nt] += c; ss[nt] += c * c; }
    }
#pragma unroll
    for (int nt = 0; nt < 8; ++nt) { red[wave][nt][lane][0] = s[nt]; red[wave][nt][lane][1] = ss[nt]; }
    __syncthreads();
    if (threadIdx.x < 128) {
        int c = threadIdx.x, nt = c >> 4, l = c & 15;
        float S = 0.f, SS = 0.f;
#pragma unroll
        for (int w = 0; w < 8; ++w) {
            S += red[w][nt][l][0] + red[w][nt][l + 16][0];
            SS += red[w][nt][l][1] + red[w][nt][l + 16][1];
        }
        unsafeAtomicAdd(&sum[c], S);
        unsafeAtomicAdd(&sumsq[c], SS);
    }
}

// ---------------- pass 2: edge MLP + scatter-add ----------------

__global__ void __launch_bounds__(256) edge_mlp_k(
    const float* __restrict__ x, const int* __restrict__ ei,
    const float* __restrict__ ea,
    const uint32_t* __restrict__ W1p, const uint32_t* __restrict__ W2p,
    const float* __restrict__ scale, const float* __restrict__ shift,
    const float* __restrict__ b2a, float* __restrict__ agg, int E, int ntiles) {
    __shared__ __align__(16) __bf16 stage[8][16][128];
    int lane = threadIdx.x & 31, wave = threadIdx.x >> 5;
    int half = lane >> 4, m = lane & 15;
    int tile = blockIdx.x * 8 + wave;
    if (tile >= ntiles) return;
    int e = tile * 16 + m;
    int r = ei[e];
    const float* xr = x + (size_t)r * 64;
    const float* er = ea + (size_t)e * 64;
    f32x8 acc[8];
#pragma unroll
    for (int nt = 0; nt < 8; ++nt) acc[nt] = zero8();
#pragma unroll
    for (int kt = 0; kt < 4; ++kt) {
        const float* src = (kt < 2) ? (xr + kt * 32) : (er + (kt - 2) * 32);
        bf16x16 a = make_afrag_g(src, half);
#pragma unroll
        for (int nt = 0; nt < 8; ++nt) {
            bf16x16 b = load_bfrag(W1p, kt * 8 + nt, lane);
            acc[nt] = wmma_bf16(a, b, acc[nt]);
        }
    }
    // BN + ReLU, stage bf16 tile (C layout: M = v + 8*half, N = nt*16 + m)
#pragma unroll
    for (int nt = 0; nt < 8; ++nt) {
        int ch = nt * 16 + m;
        float sc = scale[ch], sh = shift[ch];
#pragma unroll
        for (int v = 0; v < 8; ++v) {
            float hv = fmaf(acc[nt][v], sc, sh);
            hv = hv > 0.f ? hv : 0.f;
            stage[wave][v + 8 * half][ch] = (__bf16)hv;
        }
    }
    asm volatile("" ::: "memory");  // DS ops are in-order per wave
    const __bf16* row = &stage[wave][m][0];
    f32x8 a2[8];
#pragma unroll
    for (int nt = 0; nt < 8; ++nt) a2[nt] = zero8();
#pragma unroll
    for (int kt = 0; kt < 4; ++kt) {
        bf16x16 a = lds_afrag(row, kt, half);
#pragma unroll
        for (int nt = 0; nt < 8; ++nt) {
            bf16x16 b = load_bfrag(W2p, kt * 8 + nt, lane);
            a2[nt] = wmma_bf16(a, b, a2[nt]);
        }
    }
    int colv[8];
#pragma unroll
    for (int v = 0; v < 8; ++v) colv[v] = ei[E + tile * 16 + v + 8 * half];
#pragma unroll
    for (int nt = 0; nt < 8; ++nt) {
        float bias = b2a[nt * 16 + m];
#pragma unroll
        for (int v = 0; v < 8; ++v)
            unsafeAtomicAdd(&agg[(size_t)colv[v] * 128 + nt * 16 + m], a2[nt][v] + bias);
    }
}

// ---------------- pass 3: node BN statistics (K = 192) ----------------

__global__ void __launch_bounds__(256) node_stats_k(
    const float* __restrict__ x, const float* __restrict__ agg,
    const uint32_t* __restrict__ W1p,
    float* __restrict__ sum, float* __restrict__ sumsq, int ntiles) {
    __shared__ __align__(16) float red[8][8][32][2];
    int lane = threadIdx.x & 31, wave = threadIdx.x >> 5;
    int half = lane >> 4, m = lane & 15;
    float s[8], ss[8];
#pragma unroll
    for (int nt = 0; nt < 8; ++nt) { s[nt] = 0.f; ss[nt] = 0.f; }
    int gw = blockIdx.x * 8 + wave, nw = gridDim.x * 8;
    for (int tile = gw; tile < ntiles; tile += nw) {
        int n = tile * 16 + m;
        const float* xr = x + (size_t)n * 64;
        const float* ar = agg + (size_t)n * 128;
        f32x8 acc[8];
#pragma unroll
        for (int nt = 0; nt < 8; ++nt) acc[nt] = zero8();
#pragma unroll
        for (int kt = 0; kt < 6; ++kt) {
            const float* src = (kt < 2) ? (xr + kt * 32) : (ar + (kt - 2) * 32);
            bf16x16 a = make_afrag_g(src, half);
#pragma unroll
            for (int nt = 0; nt < 8; ++nt) {
                bf16x16 b = load_bfrag(W1p, kt * 8 + nt, lane);
                acc[nt] = wmma_bf16(a, b, acc[nt]);
            }
        }
#pragma unroll
        for (int nt = 0; nt < 8; ++nt)
#pragma unroll
            for (int v = 0; v < 8; ++v) { float c = acc[nt][v]; s[nt] += c; ss[nt] += c * c; }
    }
#pragma unroll
    for (int nt = 0; nt < 8; ++nt) { red[wave][nt][lane][0] = s[nt]; red[wave][nt][lane][1] = ss[nt]; }
    __syncthreads();
    if (threadIdx.x < 128) {
        int c = threadIdx.x, nt = c >> 4, l = c & 15;
        float S = 0.f, SS = 0.f;
#pragma unroll
        for (int w = 0; w < 8; ++w) {
            S += red[w][nt][l][0] + red[w][nt][l + 16][0];
            SS += red[w][nt][l][1] + red[w][nt][l + 16][1];
        }
        unsafeAtomicAdd(&sum[c], S);
        unsafeAtomicAdd(&sumsq[c], SS);
    }
}

// ---------------- pass 4: node MLP -> output ----------------

__global__ void __launch_bounds__(256) node_mlp_k(
    const float* __restrict__ x, const float* __restrict__ agg,
    const uint32_t* __restrict__ W1p, const uint32_t* __restrict__ W2p,
    const float* __restrict__ scale, const float* __restrict__ shift,
    const float* __restrict__ b2b, float* __restrict__ out, int ntiles) {
    __shared__ __align__(16) __bf16 stage[8][16][128];
    int lane = threadIdx.x & 31, wave = threadIdx.x >> 5;
    int half = lane >> 4, m = lane & 15;
    int tile = blockIdx.x * 8 + wave;
    if (tile >= ntiles) return;
    int n = tile * 16 + m;
    const float* xr = x + (size_t)n * 64;
    const float* ar = agg + (size_t)n * 128;
    f32x8 acc[8];
#pragma unroll
    for (int nt = 0; nt < 8; ++nt) acc[nt] = zero8();
#pragma unroll
    for (int kt = 0; kt < 6; ++kt) {
        const float* src = (kt < 2) ? (xr + kt * 32) : (ar + (kt - 2) * 32);
        bf16x16 a = make_afrag_g(src, half);
#pragma unroll
        for (int nt = 0; nt < 8; ++nt) {
            bf16x16 b = load_bfrag(W1p, kt * 8 + nt, lane);
            acc[nt] = wmma_bf16(a, b, acc[nt]);
        }
    }
#pragma unroll
    for (int nt = 0; nt < 8; ++nt) {
        int ch = nt * 16 + m;
        float sc = scale[ch], sh = shift[ch];
#pragma unroll
        for (int v = 0; v < 8; ++v) {
            float hv = fmaf(acc[nt][v], sc, sh);
            hv = hv > 0.f ? hv : 0.f;
            stage[wave][v + 8 * half][ch] = (__bf16)hv;
        }
    }
    asm volatile("" ::: "memory");
    const __bf16* row = &stage[wave][m][0];
    f32x8 a2[4];
#pragma unroll
    for (int nt = 0; nt < 4; ++nt) a2[nt] = zero8();
#pragma unroll
    for (int kt = 0; kt < 4; ++kt) {
        bf16x16 a = lds_afrag(row, kt, half);
#pragma unroll
        for (int nt = 0; nt < 4; ++nt) {
            bf16x16 b = load_bfrag(W2p, kt * 4 + nt, lane);
            a2[nt] = wmma_bf16(a, b, a2[nt]);
        }
    }
#pragma unroll
    for (int nt = 0; nt < 4; ++nt) {
        float bias = b2b[nt * 16 + m];
#pragma unroll
        for (int v = 0; v < 8; ++v) {
            int nd = tile * 16 + v + 8 * half;
            out[(size_t)nd * 64 + nt * 16 + m] = a2[nt][v] + bias;
        }
    }
}

// ---------------- host ----------------

extern "C" void kernel_launch(void* const* d_in, const int* in_sizes, int n_in,
                              void* d_out, int out_size, void* d_ws, size_t ws_size,
                              hipStream_t stream) {
    const float* x    = (const float*)d_in[0];
    const int*   ei   = (const int*)d_in[1];
    const float* ea   = (const float*)d_in[2];
    const float* W1a  = (const float*)d_in[5];
    const float* g1a  = (const float*)d_in[7];
    const float* be1a = (const float*)d_in[8];
    const float* W2a  = (const float*)d_in[9];
    const float* b2a  = (const float*)d_in[10];
    const float* W1b  = (const float*)d_in[11];
    const float* g1b  = (const float*)d_in[13];
    const float* be1b = (const float*)d_in[14];
    const float* W2b  = (const float*)d_in[15];
    const float* b2b  = (const float*)d_in[16];
    float* out = (float*)d_out;

    int N = in_sizes[0] / 64;
    int E = in_sizes[2] / 64;
    int tilesE = E / 16;
    int tilesN = N / 16;

    float* agg    = (float*)d_ws;                // N*128
    float* esum   = agg + (size_t)N * 128;       // 128
    float* esumsq = esum + 128;
    float* escale = esumsq + 128;
    float* eshift = escale + 128;
    float* nsum   = eshift + 128;
    float* nsumsq = nsum + 128;
    float* nscale = nsumsq + 128;
    float* nshift = nscale + 128;
    uint32_t* W1ap = (uint32_t*)(nshift + 128);  // 8192
    uint32_t* W2ap = W1ap + 8192;                // 8192
    uint32_t* W1bp = W2ap + 8192;                // 12288
    uint32_t* W2bp = W1bp + 12288;               // 4096

    // zero agg + stats accumulators
    zero_k<<<2048, 256, 0, stream>>>(agg, (size_t)N * 128 + 1024);

    // pre-swizzle weights into B-fragment layout (bf16 packed)
    swizzle_w_k<<<(8192 + 255) / 256, 256, 0, stream>>>(W1a, W1ap, 128, 128);
    swizzle_w_k<<<(8192 + 255) / 256, 256, 0, stream>>>(W2a, W2ap, 128, 128);
    swizzle_w_k<<<(12288 + 255) / 256, 256, 0, stream>>>(W1b, W1bp, 192, 128);
    swizzle_w_k<<<(4096 + 255) / 256, 256, 0, stream>>>(W2b, W2bp, 128, 64);

    // edge BN statistics
    edge_stats_k<<<256, 256, 0, stream>>>(x, ei, ea, W1ap, esum, esumsq, E, tilesE);
    finalize_k<<<1, 128, 0, stream>>>(esum, esumsq, g1a, be1a, escale, eshift, 1.f / (float)E, 128);

    // edge MLP + scatter-add
    edge_mlp_k<<<(tilesE + 7) / 8, 256, 0, stream>>>(x, ei, ea, W1ap, W2ap,
                                                     escale, eshift, b2a, agg, E, tilesE);

    // node BN statistics
    node_stats_k<<<128, 256, 0, stream>>>(x, agg, W1bp, nsum, nsumsq, tilesN);
    finalize_k<<<1, 128, 0, stream>>>(nsum, nsumsq, g1b, be1b, nscale, nshift, 1.f / (float)N, 128);

    // node MLP -> output
    node_mlp_k<<<(tilesN + 7) / 8, 256, 0, stream>>>(x, agg, W1bp, W2bp,
                                                     nscale, nshift, b2b, out, tilesN);
}
